// DHSRNN_62079457296497
// MI455X (gfx1250) — compile-verified, matrix operand order checked by reading, not run on hardware
//
#include <hip/hip_runtime.h>
#include <hip/hip_bf16.h>

// ---------------------------------------------------------------------------
// DH-SRNN forward on MI455X (gfx1250).
//
// Per-step dendrite GEMM: (128 x 1536) @ (1536 x 4096)  ~1.61 GFLOP
// Per-step readout GEMM:  (128 x 1024) @ (1024 x 256)   ~0.067 GFLOP
// bf16 weights: eff_W 12.6 MB + W_ro 0.5 MB -> L2-resident (192 MB L2), so
// the T=200 recurrence is matrix-core/latency bound. bf16 WMMA w/ f32 accum
// (spikes are exactly representable in bf16; x pre-converted once in prep).
//
// Data movement: all LDS tile staging uses CDNA5 async copies
// (global_load_async_to_lds_b128, ASYNCcnt) with double-buffered LDS and a
// single s_wait_asynccnt + s_barrier per K-tile. LDS rows padded to 72 bf16
// (144B = 36 banks) so 16-lane ds_load_b128 fragment reads are conflict-free.
// ---------------------------------------------------------------------------

#define T_STEPS 200
#define IN_DIM  512
#define HID     1024
#define BRANCH  4
#define NJ      (HID * BRANCH)     // 4096 dendritic rows
#define ISIZE   (IN_DIM + HID)     // 1536
#define BATCH   128
#define OUT_DIM 256
#define VTH     1.0f
#define WARMUP  10
#define LDA     72                 // padded LDS row stride (bf16 elems)

typedef __attribute__((ext_vector_type(16))) __bf16 v16bf;
typedef __attribute__((ext_vector_type(8)))  float  v8f;

__device__ __forceinline__ float sigmoidf_dev(float v) {
    return 1.0f / (1.0f + __expf(-v));
}

// Async 16B copy global -> LDS (VDST = LDS byte address, VADDR = 64b global).
// Tracked by ASYNCcnt; completion enforced with s_wait_asynccnt.
__device__ __forceinline__ void async_b128(const __hip_bfloat16* gsrc,
                                           __hip_bfloat16* ldst) {
    asm volatile("global_load_async_to_lds_b128 %0, %1, off"
                 :: "v"((unsigned int)(size_t)ldst), "v"(gsrc)
                 : "memory");
}
__device__ __forceinline__ void async_wait0() {
    asm volatile("s_wait_asynccnt 0x0" ::: "memory");
}

// ---------------------------------------------------------------------------
// Prep: x -> bf16 copy, eff_W = W_dense*mask -> bf16, W_ro -> bf16,
// decay sigmoids, zero all state. One element per thread.
// ---------------------------------------------------------------------------
__global__ void dhsrnn_prep(const float* __restrict__ x,
                            const float* __restrict__ Wd,
                            const float* __restrict__ mask,
                            const float* __restrict__ Wro,
                            const float* __restrict__ tau_n,
                            const float* __restrict__ tau_m,
                            const float* __restrict__ tau_ro,
                            __hip_bfloat16* __restrict__ xb,
                            __hip_bfloat16* __restrict__ effW,
                            __hip_bfloat16* __restrict__ WroB,
                            float* __restrict__ beta,
                            float* __restrict__ alpha,
                            float* __restrict__ alpha_o,
                            float* __restrict__ d_state,
                            float* __restrict__ mem,
                            __hip_bfloat16* __restrict__ spk0,
                            __hip_bfloat16* __restrict__ spk1,
                            float* __restrict__ mem_o,
                            float* __restrict__ accum)
{
    size_t id = (size_t)blockIdx.x * blockDim.x + threadIdx.x;

    const size_t N_XB = (size_t)BATCH * T_STEPS * IN_DIM;   // 13107200
    if (id < N_XB) { xb[id] = __float2bfloat16(x[id]); return; }
    id -= N_XB;
    const size_t N_EFFW = (size_t)NJ * ISIZE;               // 6291456
    if (id < N_EFFW) { effW[id] = __float2bfloat16(Wd[id] * mask[id]); return; }
    id -= N_EFFW;
    const size_t N_WRO = (size_t)OUT_DIM * HID;             // 262144
    if (id < N_WRO) { WroB[id] = __float2bfloat16(Wro[id]); return; }
    id -= N_WRO;
    if (id < NJ)      { beta[id]    = sigmoidf_dev(tau_n[id]);  return; }
    id -= NJ;
    if (id < HID)     { alpha[id]   = sigmoidf_dev(tau_m[id]);  return; }
    id -= HID;
    if (id < OUT_DIM) { alpha_o[id] = sigmoidf_dev(tau_ro[id]); return; }
    id -= OUT_DIM;
    const size_t N_D = (size_t)BATCH * NJ;                  // 524288
    if (id < N_D) { d_state[id] = 0.0f; return; }
    id -= N_D;
    const size_t N_M = (size_t)BATCH * HID;                 // 131072
    if (id < N_M) { mem[id] = 0.0f; return; }
    id -= N_M;
    if (id < N_M) { spk0[id] = __float2bfloat16(0.0f); return; }
    id -= N_M;
    if (id < N_M) { spk1[id] = __float2bfloat16(0.0f); return; }
    id -= N_M;
    const size_t N_O = (size_t)BATCH * OUT_DIM;             // 32768
    if (id < N_O) { mem_o[id] = 0.0f; return; }
    id -= N_O;
    if (id < N_O) { accum[id] = 0.0f; return; }
}

// ---------------------------------------------------------------------------
// Dendrite step: one WG = 256 threads (8 waves), all 128 batch rows, 64-wide
// j tile (16 complete hidden neurons -> branch sum fused in epilogue).
// K loop: 24 iterations of K-tile 64 (8 WMMAs/wave/iter), async double buffer.
// ---------------------------------------------------------------------------
__global__ __launch_bounds__(256)
void dhsrnn_dend_step(const __hip_bfloat16* __restrict__ xb,
                      const __hip_bfloat16* __restrict__ effW,
                      const float* __restrict__ b_dense,
                      const float* __restrict__ beta,
                      const float* __restrict__ alpha,
                      float* __restrict__ d_state,
                      float* __restrict__ mem,
                      const __hip_bfloat16* __restrict__ spk_in,
                      __hip_bfloat16* __restrict__ spk_out,
                      int t)
{
    __shared__ alignas(64) __hip_bfloat16 lA[2][BATCH][LDA]; // 36.9 KB
    __shared__ alignas(64) __hip_bfloat16 lB[2][64][LDA];    // 18.4 KB

    const int tid   = threadIdx.x;
    const int lane  = tid & 31;
    const int wave  = tid >> 5;
    const int jbase = blockIdx.x * 64;

    const v8f vzero = {0.f, 0.f, 0.f, 0.f, 0.f, 0.f, 0.f, 0.f};
    v8f acc[4];
    acc[0] = vzero; acc[1] = vzero; acc[2] = vzero; acc[3] = vzero;

    // A staging: row = tid/2, 4 chunks of 8 bf16 (64B contiguous global)
    const int ar  = tid >> 1;
    const int ac0 = (tid & 1) * 32;          // bf16 offset of first chunk
    // B staging: row = tid/4, 2 chunks of 8 bf16 (32B contiguous global)
    const int br  = tid >> 2;
    const int bc0 = (tid & 3) * 16;

    auto stage = [&](int sbuf, int k0) {
        const __hip_bfloat16* asrc =
            (k0 < IN_DIM)
                ? xb + ((size_t)ar * T_STEPS + t) * IN_DIM + k0 + ac0
                : spk_in + (size_t)ar * HID + (k0 - IN_DIM) + ac0;
        __hip_bfloat16* adst = &lA[sbuf][ar][ac0];
#pragma unroll
        for (int c = 0; c < 4; ++c)
            async_b128(asrc + c * 8, adst + c * 8);

        const __hip_bfloat16* bsrc =
            effW + (size_t)(jbase + br) * ISIZE + k0 + bc0;
        __hip_bfloat16* bdst = &lB[sbuf][br][bc0];
#pragma unroll
        for (int c = 0; c < 2; ++c)
            async_b128(bsrc + c * 8, bdst + c * 8);
    };

    auto compute = [&](int cbuf) {
#pragma unroll
        for (int kk = 0; kk < 2; ++kk) {
            v16bf a = *reinterpret_cast<const v16bf*>(
                &lA[cbuf][wave * 16 + (lane & 15)][kk * 32 + (lane >> 4) * 16]);
#pragma unroll
            for (int s = 0; s < 4; ++s) {
                v16bf bf = *reinterpret_cast<const v16bf*>(
                    &lB[cbuf][s * 16 + (lane & 15)][kk * 32 + (lane >> 4) * 16]);
                acc[s] = __builtin_amdgcn_wmma_f32_16x16x32_bf16(
                    false, a, false, bf, (short)0, acc[s], false, false);
            }
        }
    };

    constexpr int KT = ISIZE / 64;   // 24
    stage(0, 0);
    int buf = 0;
    for (int it = 0; it < KT; ++it) {
        async_wait0();               // my async writes to lX[buf] landed
        __syncthreads();             // everyone's landed; prev readers of buf^1 done
        if (it + 1 < KT) stage(buf ^ 1, (it + 1) * 64);
        compute(buf);
        buf ^= 1;
    }

    // --- epilogue: d update, branch sum via shfl over 4 branch lanes, soma ---
#pragma unroll
    for (int s = 0; s < 4; ++s) {
#pragma unroll
        for (int r = 0; r < 8; ++r) {
            const int b = wave * 16 + r + ((lane >> 4) << 3); // M = r + 8*(lane/16)
            const int j = jbase + s * 16 + (lane & 15);        // N = lane&15
            const float di  = acc[s][r] + b_dense[j];
            const float bet = beta[j];
            const size_t dix = (size_t)b * NJ + j;
            const float dn = bet * d_state[dix] + (1.0f - bet) * di;
            d_state[dix] = dn;
            float sum = dn;
            sum += __shfl_xor(sum, 1, 32);
            sum += __shfl_xor(sum, 2, 32);
            if ((lane & 3) == 0) {
                const int i = j >> 2;
                const size_t mix = (size_t)b * HID + i;
                const float al   = alpha[i];
                const float spko = __bfloat162float(spk_in[mix]);
                const float mn   = al * mem[mix] + (1.0f - al) * sum - VTH * spko;
                mem[mix] = mn;
                spk_out[mix] = __float2bfloat16(mn > VTH ? 1.0f : 0.0f);
            }
        }
    }
}

// ---------------------------------------------------------------------------
// Readout step: spk @ W_ro^T + b_ro, leaky integrate, accumulate mean (t>=10).
// Grid 4 (OUT tiles of 64); 16 iterations of K-tile 64, same async pipeline.
// ---------------------------------------------------------------------------
__global__ __launch_bounds__(256)
void dhsrnn_readout_step(const __hip_bfloat16* __restrict__ spk,
                         const __hip_bfloat16* __restrict__ WroB,
                         const float* __restrict__ b_ro,
                         const float* __restrict__ alpha_o,
                         float* __restrict__ mem_o,
                         float* __restrict__ accum,
                         int t)
{
    __shared__ alignas(64) __hip_bfloat16 lA[2][BATCH][LDA];
    __shared__ alignas(64) __hip_bfloat16 lB[2][64][LDA];

    const int tid   = threadIdx.x;
    const int lane  = tid & 31;
    const int wave  = tid >> 5;
    const int obase = blockIdx.x * 64;

    const v8f vzero = {0.f, 0.f, 0.f, 0.f, 0.f, 0.f, 0.f, 0.f};
    v8f acc[4];
    acc[0] = vzero; acc[1] = vzero; acc[2] = vzero; acc[3] = vzero;

    const int ar  = tid >> 1;
    const int ac0 = (tid & 1) * 32;
    const int br  = tid >> 2;
    const int bc0 = (tid & 3) * 16;

    auto stage = [&](int sbuf, int k0) {
        const __hip_bfloat16* asrc = spk + (size_t)ar * HID + k0 + ac0;
        __hip_bfloat16* adst = &lA[sbuf][ar][ac0];
#pragma unroll
        for (int c = 0; c < 4; ++c)
            async_b128(asrc + c * 8, adst + c * 8);

        const __hip_bfloat16* bsrc =
            WroB + (size_t)(obase + br) * HID + k0 + bc0;
        __hip_bfloat16* bdst = &lB[sbuf][br][bc0];
#pragma unroll
        for (int c = 0; c < 2; ++c)
            async_b128(bsrc + c * 8, bdst + c * 8);
    };

    auto compute = [&](int cbuf) {
#pragma unroll
        for (int kk = 0; kk < 2; ++kk) {
            v16bf a = *reinterpret_cast<const v16bf*>(
                &lA[cbuf][wave * 16 + (lane & 15)][kk * 32 + (lane >> 4) * 16]);
#pragma unroll
            for (int s = 0; s < 4; ++s) {
                v16bf bf = *reinterpret_cast<const v16bf*>(
                    &lB[cbuf][s * 16 + (lane & 15)][kk * 32 + (lane >> 4) * 16]);
                acc[s] = __builtin_amdgcn_wmma_f32_16x16x32_bf16(
                    false, a, false, bf, (short)0, acc[s], false, false);
            }
        }
    };

    constexpr int KT = HID / 64;     // 16
    stage(0, 0);
    int buf = 0;
    for (int it = 0; it < KT; ++it) {
        async_wait0();
        __syncthreads();
        if (it + 1 < KT) stage(buf ^ 1, (it + 1) * 64);
        compute(buf);
        buf ^= 1;
    }

#pragma unroll
    for (int s = 0; s < 4; ++s) {
#pragma unroll
        for (int r = 0; r < 8; ++r) {
            const int b = wave * 16 + r + ((lane >> 4) << 3);
            const int o = obase + s * 16 + (lane & 15);
            const float v  = acc[s][r] + b_ro[o];
            const float ao = alpha_o[o];
            const size_t ix = (size_t)b * OUT_DIM + o;
            const float mo = ao * mem_o[ix] + (1.0f - ao) * v;
            mem_o[ix] = mo;
            if (t >= WARMUP) accum[ix] += mo;
        }
    }
}

__global__ void dhsrnn_finalize(const float* __restrict__ accum,
                                float* __restrict__ out)
{
    const int id = blockIdx.x * blockDim.x + threadIdx.x;
    if (id < BATCH * OUT_DIM)
        out[id] = accum[id] * (1.0f / (float)(T_STEPS - WARMUP));
}

// ---------------------------------------------------------------------------
// Host launcher: prep, 200 x (dend + readout) stream-ordered, finalize.
// Workspace carved from d_ws (~43 MB), 256B aligned.
// ---------------------------------------------------------------------------
extern "C" void kernel_launch(void* const* d_in, const int* in_sizes, int n_in,
                              void* d_out, int out_size, void* d_ws, size_t ws_size,
                              hipStream_t stream)
{
    const float* x      = (const float*)d_in[0];
    const float* Wd     = (const float*)d_in[1];
    const float* bden   = (const float*)d_in[2];
    const float* mask   = (const float*)d_in[3];
    const float* tau_n  = (const float*)d_in[4];
    const float* tau_m  = (const float*)d_in[5];
    const float* Wro    = (const float*)d_in[6];
    const float* bro    = (const float*)d_in[7];
    const float* tauro  = (const float*)d_in[8];

    char* p = (char*)d_ws;
    auto carve = [&](size_t bytes) {
        char* r = p;
        p += (bytes + 255) & ~(size_t)255;
        return r;
    };
    __hip_bfloat16* xb    = (__hip_bfloat16*)carve((size_t)BATCH * T_STEPS * IN_DIM * 2);
    __hip_bfloat16* effW  = (__hip_bfloat16*)carve((size_t)NJ * ISIZE * 2);
    __hip_bfloat16* WroB  = (__hip_bfloat16*)carve((size_t)OUT_DIM * HID * 2);
    float* beta    = (float*)carve((size_t)NJ * 4);
    float* alpha   = (float*)carve((size_t)HID * 4);
    float* alpha_o = (float*)carve((size_t)OUT_DIM * 4);
    float* d_state = (float*)carve((size_t)BATCH * NJ * 4);
    float* mem     = (float*)carve((size_t)BATCH * HID * 4);
    __hip_bfloat16* spk0 = (__hip_bfloat16*)carve((size_t)BATCH * HID * 2);
    __hip_bfloat16* spk1 = (__hip_bfloat16*)carve((size_t)BATCH * HID * 2);
    float* mem_o   = (float*)carve((size_t)BATCH * OUT_DIM * 4);
    float* accum   = (float*)carve((size_t)BATCH * OUT_DIM * 4);

    const size_t prep_total =
        (size_t)BATCH * T_STEPS * IN_DIM +
        (size_t)NJ * ISIZE + (size_t)OUT_DIM * HID + NJ + HID + OUT_DIM +
        (size_t)BATCH * NJ + 3 * (size_t)BATCH * HID + 2 * (size_t)BATCH * OUT_DIM;
    const int prep_blocks = (int)((prep_total + 255) / 256);
    dhsrnn_prep<<<prep_blocks, 256, 0, stream>>>(
        x, Wd, mask, Wro, tau_n, tau_m, tauro,
        xb, effW, WroB, beta, alpha, alpha_o,
        d_state, mem, spk0, spk1, mem_o, accum);

    for (int t = 0; t < T_STEPS; ++t) {
        const __hip_bfloat16* spk_in  = (t & 1) ? spk1 : spk0;
        __hip_bfloat16*       spk_out = (t & 1) ? spk0 : spk1;
        dhsrnn_dend_step<<<dim3(NJ / 64), 256, 0, stream>>>(
            xb, effW, bden, beta, alpha, d_state, mem, spk_in, spk_out, t);
        dhsrnn_readout_step<<<dim3(OUT_DIM / 64), 256, 0, stream>>>(
            spk_out, WroB, bro, alpha_o, mem_o, accum, t);
    }

    dhsrnn_finalize<<<dim3((BATCH * OUT_DIM) / 256), 256, 0, stream>>>(
        accum, (float*)d_out);
}
